// SelfAttention_69956427317645
// MI455X (gfx1250) — matrix operand order repaired and checked by reading, compiled, not run on hardware
//
#include <hip/hip_runtime.h>
#include <stdint.h>

// ---------------------------------------------------------------------------
// Self-attention block for MI455X (gfx1250, wave32, WMMA bf16 16x16x32).
// B=8, N=2048, C=512. Flash-attention fusion: scores never hit HBM.
// Data movement uses CDNA5 async DMA (GLOBAL_LOAD_ASYNC_TO_LDS_B128) with
// double-buffered LDS tiles overlapped against the WMMA pipe.
// ---------------------------------------------------------------------------

#define CCH   512
#define NSEQ  2048
#define NBAT  8
#define NROWS (NBAT * NSEQ)          // 16384
#define NEL   (NROWS * CCH)          // 8388608

typedef __attribute__((ext_vector_type(16))) __bf16 v16bf;
typedef __attribute__((ext_vector_type(8)))  float  v8f;

union ABu { uint4 u[2]; v16bf v; };  // 32B WMMA A/B operand (8 VGPRs)

__device__ inline unsigned short f2b(float f) {      // f32 -> bf16 (RNE)
  unsigned int u = __float_as_uint(f);
  u += 0x7fffu + ((u >> 16) & 1u);
  return (unsigned short)(u >> 16);
}

__device__ inline v8f wmma_bf16(const ABu& a, const ABu& b, v8f c) {
  // D(16x16 f32) = A(16x32 bf16) * B(32x16 bf16) + C
  return __builtin_amdgcn_wmma_f32_16x16x32_bf16(
      /*neg_a=*/false, a.v, /*neg_b=*/false, b.v,
      /*c_mod=*/(short)0, c, /*reuse_a=*/false, /*reuse_b=*/false);
}

// CDNA5 async DMA: 16B memory -> LDS per enabled lane, tracked by ASYNCcnt.
// Flat/generic LDS pointers carry the LDS byte offset in bits[31:0], which is
// exactly the VDST operand.
__device__ __forceinline__ void async_ld_b128(void* lds, const void* g) {
  asm volatile("global_load_async_to_lds_b128 %0, %1, off"
               :
               : "v"((uint32_t)(uintptr_t)lds), "v"((uint64_t)(uintptr_t)g)
               : "memory");
}

// ---------------------------------------------------------------------------
// Precision converts. Weights are stored transposed (WT[n][k] = W[k][n]) so
// that WMMA B-operand "columns" become contiguous rows in memory/LDS.
// ---------------------------------------------------------------------------
__global__ __launch_bounds__(256) void cvt_f32_bf16(
    const float* __restrict__ in, unsigned short* __restrict__ out, int n) {
  int i = blockIdx.x * 256 + threadIdx.x;
  if (i < n) out[i] = f2b(in[i]);
}

__global__ __launch_bounds__(256) void cvt_wT(
    const float* __restrict__ W, unsigned short* __restrict__ WT) {
  int i = blockIdx.x * 256 + threadIdx.x;   // i = n*512 + k
  int n = i >> 9, k = i & 511;
  WT[i] = f2b(W[(k << 9) + n]);
}

// ---------------------------------------------------------------------------
// Projection GEMM:  out[R x 512] = A[R x 512] * W + bias
//   WG = 128 rows x 64 cols, 8 waves; wave owns 16 rows x 64 cols (4 D-tiles).
//   Double-buffered async-DMA staging of A and W tiles.
//   mode 0: bf16 row-major out (f, g)
//   mode 1: bf16 transposed out [B][C][N] (h, feeds the P*h B-operand)
//   mode 2: f32 out + fused per-channel sum / sum-of-squares (BN stats)
// ---------------------------------------------------------------------------
__global__ __launch_bounds__(256) void proj_gemm(
    const unsigned short* __restrict__ A,    // [R][512] bf16
    const unsigned short* __restrict__ WT,   // [512][512] bf16, WT[n][k]
    const float* __restrict__ bias,          // [512]
    unsigned short* __restrict__ outb,
    float* __restrict__ outf,
    float* __restrict__ sums,                // [2*512]
    int mode) {
  __shared__ __attribute__((aligned(16))) unsigned short xs[2][128 * 32]; // 16KB
  __shared__ __attribute__((aligned(16))) unsigned short wsh[2][64 * 32]; //  8KB

  const int tid  = threadIdx.x;
  const int lane = tid & 31, wv = tid >> 5;
  const int half = lane >> 4, ln = lane & 15;
  const int row0 = blockIdx.x * 128;
  const int col0 = blockIdx.y * 64;

  // Issue the 3 async b128 DMAs per thread that fill tile buffer (kt&1).
  auto issue_tiles = [&](int kt) {
    const int buf = kt & 1;
    const int k0 = kt * 32;
#pragma unroll
    for (int i = tid; i < 512; i += 256) {        // A tile 128x32
      int r = i >> 2, q = i & 3;
      async_ld_b128(&xs[buf][(r << 5) + (q << 3)],
                    &A[(size_t)(row0 + r) * CCH + k0 + (q << 3)]);
    }
    {                                             // W tile 64x32
      int n = tid >> 2, q = tid & 3;
      async_ld_b128(&wsh[buf][(n << 5) + (q << 3)],
                    &WT[(size_t)(col0 + n) * CCH + k0 + (q << 3)]);
    }
  };

  v8f zero = {};
  v8f acc[4];
#pragma unroll
  for (int t = 0; t < 4; ++t) acc[t] = zero;

  issue_tiles(0);
  for (int kt = 0; kt < 16; ++kt) {
    const int buf = kt & 1;
    __syncthreads();                       // everyone done reading buf^1
    if (kt < 15) {
      issue_tiles(kt + 1);                 // prefetch next tile (DMA overlap)
      asm volatile("s_wait_asynccnt 0x3" ::: "memory");  // oldest 3 (=current) done
    } else {
      asm volatile("s_wait_asynccnt 0x0" ::: "memory");
    }
    __syncthreads();                       // current tile globally visible

    ABu a;  // A-operand: lane half owns K {half*8..+7, 16+half*8..+7}
    const int arow = wv * 16 + ln;
    a.u[0] = *(const uint4*)&xs[buf][(arow << 5) + (half << 3)];
    a.u[1] = *(const uint4*)&xs[buf][(arow << 5) + 16 + (half << 3)];
#pragma unroll
    for (int t = 0; t < 4; ++t) {
      ABu b;  // B-operand: lane half owns K {half*8..+7, 16+half*8..+7}, col = ln
      const int brow = t * 16 + ln;
      b.u[0] = *(const uint4*)&wsh[buf][(brow << 5) + (half << 3)];
      b.u[1] = *(const uint4*)&wsh[buf][(brow << 5) + 16 + (half << 3)];
      acc[t] = wmma_bf16(a, b, acc[t]);
    }
  }

  // epilogue — D layout: row M = r + 8*half, col = ln
#pragma unroll
  for (int t = 0; t < 4; ++t) {
    const int cc = col0 + t * 16 + ln;
    const float bs = bias[cc];
    if (mode == 2) {
      float csum = 0.f, csq = 0.f;
#pragma unroll
      for (int r = 0; r < 8; ++r) {
        const int row = row0 + wv * 16 + r + half * 8;
        float y = acc[t][r] + bs;
        outf[(size_t)row * CCH + cc] = y;
        csum += y; csq += y * y;
      }
      csum += __shfl_xor(csum, 16, 32);
      csq  += __shfl_xor(csq, 16, 32);
      if (lane < 16) {
        atomicAdd(&sums[cc], csum);
        atomicAdd(&sums[CCH + cc], csq);
      }
    } else if (mode == 1) {
#pragma unroll
      for (int r = 0; r < 8; ++r) {
        const int row = row0 + wv * 16 + r + half * 8;
        const int bt = row >> 11;          // batch
        const int n  = row & 2047;
        outb[((size_t)(bt * CCH + cc) << 11) + n] = f2b(acc[t][r] + bs);
      }
    } else {
#pragma unroll
      for (int r = 0; r < 8; ++r) {
        const int row = row0 + wv * 16 + r + half * 8;
        outb[(size_t)row * CCH + cc] = f2b(acc[t][r] + bs);
      }
    }
  }
}

// ---------------------------------------------------------------------------
// Flash attention: WG = (16 query rows, full C). 4 waves; wave w owns
// C-slice [w*128, w*128+128) for both the score K-reduction (partials reduced
// through LDS) and the P*h output accumulation. KV blocks of 32, with
// double-buffered async-DMA staging of g / h^T blocks.
// ---------------------------------------------------------------------------
__global__ __launch_bounds__(128) void flash_attn(
    const unsigned short* __restrict__ fB,   // [B*N][C] bf16
    const unsigned short* __restrict__ gB,   // [B*N][C] bf16
    const unsigned short* __restrict__ hT,   // [B][C][N] bf16
    unsigned short* __restrict__ oB) {       // [B*N][C] bf16
  __shared__ __attribute__((aligned(16))) unsigned short fs[16 * CCH];     // 16KB
  __shared__ __attribute__((aligned(16))) unsigned short gs[2][32 * CCH];  // 64KB
  __shared__ __attribute__((aligned(16))) unsigned short hs[2][CCH * 32];  // 64KB
  __shared__ __attribute__((aligned(16))) float sred[4 * 2 * 256];         //  8KB
  __shared__ __attribute__((aligned(16))) unsigned short pscr[4 * 512];    //  4KB

  const int tid  = threadIdx.x;
  const int lane = tid & 31, wv = tid >> 5;
  const int half = lane >> 4, ln = lane & 15;
  const int qb = blockIdx.x, bt = blockIdx.y;
  const int kbase = wv * 128;

  // Issue the 32 async b128 DMAs per thread filling g/h buffer (j&1).
  auto issue_gh = [&](int j) {
    const int buf = j & 1;
    const size_t gbase = ((size_t)bt * NSEQ + j * 32) * CCH;
    for (int i = tid; i < 2048; i += 128)           // g block 32x512 (flat)
      async_ld_b128(&gs[buf][i << 3], &gB[gbase + ((size_t)i << 3)]);
    const size_t hbase = (size_t)bt * CCH * NSEQ + (size_t)j * 32;
    for (int i = tid; i < 2048; i += 128) {         // hT block 512x32 (strided)
      int c = i >> 2, q = i & 3;
      async_ld_b128(&hs[buf][(c << 5) + (q << 3)],
                    &hT[hbase + (size_t)c * NSEQ + (q << 3)]);
    }
  };

  // f block (16 rows x 512) once, via async DMA as well (8 per thread).
  const size_t fbase = ((size_t)bt * NSEQ + qb * 16) * CCH;
  for (int i = tid; i < 1024; i += 128)
    async_ld_b128(&fs[i << 3], &fB[fbase + ((size_t)i << 3)]);
  issue_gh(0);

  v8f zero = {};
  v8f oacc[8];
#pragma unroll
  for (int t = 0; t < 8; ++t) oacc[t] = zero;
  float m[8], l[8];
#pragma unroll
  for (int r = 0; r < 8; ++r) { m[r] = -3.0e30f; l[r] = 0.f; }

  for (int j = 0; j < 64; ++j) {
    const int buf = j & 1;
    __syncthreads();                 // all waves done reading buf^1
    if (j < 63) {
      issue_gh(j + 1);               // prefetch next KV block (DMA overlap)
      // oldest 32 outstanding = current buffer (+ f on j==0): wait them out
      asm volatile("s_wait_asynccnt 0x20" ::: "memory");
    } else {
      asm volatile("s_wait_asynccnt 0x0" ::: "memory");
    }
    __syncthreads();                 // current buffer globally visible

    // partial scores S = f * g^T over this wave's 128-wide C slice
    v8f s0 = zero, s1 = zero;
#pragma unroll
    for (int kk = 0; kk < 4; ++kk) {
      const int k0 = kbase + kk * 32;
      ABu a, b0, b1;
      a.u[0]  = *(const uint4*)&fs[(ln << 9) + k0 + (half << 3)];
      a.u[1]  = *(const uint4*)&fs[(ln << 9) + k0 + 16 + (half << 3)];
      b0.u[0] = *(const uint4*)&gs[buf][(ln << 9) + k0 + (half << 3)];
      b0.u[1] = *(const uint4*)&gs[buf][(ln << 9) + k0 + 16 + (half << 3)];
      s0 = wmma_bf16(a, b0, s0);
      b1.u[0] = *(const uint4*)&gs[buf][((16 + ln) << 9) + k0 + (half << 3)];
      b1.u[1] = *(const uint4*)&gs[buf][((16 + ln) << 9) + k0 + 16 + (half << 3)];
      s1 = wmma_bf16(a, b1, s1);
    }
    // cross-wave K reduction of the 16x32 score tile
    {
      float* p0 = &sred[(wv * 2 + 0) * 256 + lane * 8];
      float* p1 = &sred[(wv * 2 + 1) * 256 + lane * 8];
#pragma unroll
      for (int r = 0; r < 8; ++r) { p0[r] = s0[r]; p1[r] = s1[r]; }
    }
    __syncthreads();
    float t0[8], t1[8];
#pragma unroll
    for (int r = 0; r < 8; ++r) { t0[r] = 0.f; t1[r] = 0.f; }
#pragma unroll
    for (int w2 = 0; w2 < 4; ++w2) {
      const float* p0 = &sred[(w2 * 2 + 0) * 256 + lane * 8];
      const float* p1 = &sred[(w2 * 2 + 1) * 256 + lane * 8];
#pragma unroll
      for (int r = 0; r < 8; ++r) { t0[r] += p0[r]; t1[r] += p1[r]; }
    }

    // online softmax (f32). Row M = r + 8*half lives on 16 lanes of one half.
    float alpha[8], p0v[8], p1v[8];
#pragma unroll
    for (int r = 0; r < 8; ++r) {
      float rm = fmaxf(t0[r], t1[r]);
      rm = fmaxf(rm, __shfl_xor(rm, 1, 32));
      rm = fmaxf(rm, __shfl_xor(rm, 2, 32));
      rm = fmaxf(rm, __shfl_xor(rm, 4, 32));
      rm = fmaxf(rm, __shfl_xor(rm, 8, 32));
      const float nm = fmaxf(m[r], rm);
      alpha[r] = __expf(m[r] - nm);
      m[r] = nm;
      p0v[r] = __expf(t0[r] - nm);
      p1v[r] = __expf(t1[r] - nm);
      float s = p0v[r] + p1v[r];
      s += __shfl_xor(s, 1, 32);
      s += __shfl_xor(s, 2, 32);
      s += __shfl_xor(s, 4, 32);
      s += __shfl_xor(s, 8, 32);
      l[r] = l[r] * alpha[r] + s;
    }
#pragma unroll
    for (int t = 0; t < 8; ++t)
#pragma unroll
      for (int r = 0; r < 8; ++r) oacc[t][r] *= alpha[r];

    // repack P (D layout) -> A-operand layout via per-wave LDS scratch
    unsigned short* ps = &pscr[wv * 512];
#pragma unroll
    for (int r = 0; r < 8; ++r) {
      const int M = r + half * 8;
      ps[(M << 5) + ln]      = f2b(p0v[r]);
      ps[(M << 5) + 16 + ln] = f2b(p1v[r]);
    }
    asm volatile("s_wait_dscnt 0" ::: "memory");  // intra-wave LDS RAW fence
    ABu pa;
    pa.u[0] = *(const uint4*)&ps[(ln << 5) + (half << 3)];
    pa.u[1] = *(const uint4*)&ps[(ln << 5) + 16 + (half << 3)];

    // O += P * h for this wave's 8 C-chunks (h^T makes B columns contiguous)
#pragma unroll
    for (int t = 0; t < 8; ++t) {
      const int c0 = kbase + t * 16;
      ABu b;
      b.u[0] = *(const uint4*)&hs[buf][((c0 + ln) << 5) + (half << 3)];
      b.u[1] = *(const uint4*)&hs[buf][((c0 + ln) << 5) + 16 + (half << 3)];
      oacc[t] = wmma_bf16(pa, b, oacc[t]);
    }
  }

  // finalize: divide by row sums, store bf16
#pragma unroll
  for (int r = 0; r < 8; ++r) l[r] = 1.0f / l[r];
  const size_t orow0 = (size_t)bt * NSEQ + qb * 16;
#pragma unroll
  for (int t = 0; t < 8; ++t) {
    const int cc = kbase + t * 16 + ln;
#pragma unroll
    for (int r = 0; r < 8; ++r) {
      const size_t row = orow0 + r + half * 8;
      oB[row * CCH + cc] = f2b(oacc[t][r] * l[r]);
    }
  }
}

// ---------------------------------------------------------------------------
// BatchNorm finalize + apply (+ residual)
// ---------------------------------------------------------------------------
__global__ void bn_stats(const float* __restrict__ sums,
                         const float* __restrict__ gamma,
                         const float* __restrict__ beta,
                         float* __restrict__ ss) {
  int c = threadIdx.x;
  const float inv_n = 1.0f / (float)NROWS;
  float mean = sums[c] * inv_n;
  float var  = sums[CCH + c] * inv_n - mean * mean;
  float sc   = gamma[c] * rsqrtf(var + 1e-3f);
  ss[c]       = sc;
  ss[CCH + c] = beta[c] - mean * sc;
}

__global__ __launch_bounds__(256) void bn_apply(
    const float* __restrict__ y, const float* __restrict__ ss,
    const float* __restrict__ x, float* __restrict__ out, int n4) {
  int i = blockIdx.x * 256 + threadIdx.x;
  if (i >= n4) return;
  const int c4 = (i & 127) << 2;          // channel base (C=512 -> 128 vec4)
  float4 yv = ((const float4*)y)[i];
  float4 xv = ((const float4*)x)[i];
  float4 sc = *(const float4*)&ss[c4];
  float4 sh = *(const float4*)&ss[CCH + c4];
  float4 o;
  o.x = yv.x * sc.x + sh.x + xv.x;
  o.y = yv.y * sc.y + sh.y + xv.y;
  o.z = yv.z * sc.z + sh.z + xv.z;
  o.w = yv.w * sc.w + sh.w + xv.w;
  ((float4*)out)[i] = o;
}

// ---------------------------------------------------------------------------
extern "C" void kernel_launch(void* const* d_in, const int* in_sizes, int n_in,
                              void* d_out, int out_size, void* d_ws,
                              size_t ws_size, hipStream_t stream) {
  (void)in_sizes; (void)n_in; (void)out_size; (void)ws_size;
  const float* x     = (const float*)d_in[0];
  const float* Wf    = (const float*)d_in[1];
  const float* bf    = (const float*)d_in[2];
  const float* Wg    = (const float*)d_in[3];
  const float* bg    = (const float*)d_in[4];
  const float* Wh    = (const float*)d_in[5];
  const float* bh    = (const float*)d_in[6];
  const float* Wo    = (const float*)d_in[7];
  const float* bo    = (const float*)d_in[8];
  const float* gamma = (const float*)d_in[9];
  const float* beta  = (const float*)d_in[10];

  char* ws = (char*)d_ws;
  size_t cur = 0;
  auto alloc = [&](size_t bytes) -> char* {
    char* p = ws + cur;
    cur = (cur + bytes + 255) & ~(size_t)255;
    return p;
  };
  unsigned short* xbf = (unsigned short*)alloc((size_t)NEL * 2);
  unsigned short* wfT = (unsigned short*)alloc((size_t)CCH * CCH * 2);
  unsigned short* wgT = (unsigned short*)alloc((size_t)CCH * CCH * 2);
  unsigned short* whT = (unsigned short*)alloc((size_t)CCH * CCH * 2);
  unsigned short* woT = (unsigned short*)alloc((size_t)CCH * CCH * 2);
  unsigned short* fb  = (unsigned short*)alloc((size_t)NEL * 2);
  unsigned short* gb  = (unsigned short*)alloc((size_t)NEL * 2);
  unsigned short* htb = (unsigned short*)alloc((size_t)NEL * 2);
  unsigned short* ob  = (unsigned short*)alloc((size_t)NEL * 2);
  float* yb   = (float*)alloc((size_t)NEL * 4);
  float* sums = (float*)alloc((size_t)2 * CCH * 4);
  float* ssb  = (float*)alloc((size_t)2 * CCH * 4);

  // 1) precision converts
  cvt_f32_bf16<<<NEL / 256, 256, 0, stream>>>(x, xbf, NEL);
  cvt_wT<<<(CCH * CCH) / 256, 256, 0, stream>>>(Wf, wfT);
  cvt_wT<<<(CCH * CCH) / 256, 256, 0, stream>>>(Wg, wgT);
  cvt_wT<<<(CCH * CCH) / 256, 256, 0, stream>>>(Wh, whT);
  cvt_wT<<<(CCH * CCH) / 256, 256, 0, stream>>>(Wo, woT);

  // 2) QKV projections (h transposed for the P*h B-operand)
  dim3 pg(NROWS / 128, CCH / 64);
  proj_gemm<<<pg, 256, 0, stream>>>(xbf, wfT, bf, fb, nullptr, nullptr, 0);
  proj_gemm<<<pg, 256, 0, stream>>>(xbf, wgT, bg, gb, nullptr, nullptr, 0);
  proj_gemm<<<pg, 256, 0, stream>>>(xbf, whT, bh, htb, nullptr, nullptr, 1);

  // 3) flash attention
  flash_attn<<<dim3(NSEQ / 16, NBAT), 128, 0, stream>>>(fb, gb, htb, ob);

  // 4) output projection + fused BN statistics
  hipMemsetAsync(sums, 0, 2 * CCH * sizeof(float), stream);
  proj_gemm<<<pg, 256, 0, stream>>>(ob, woT, bo, nullptr, yb, sums, 2);

  // 5) BN finalize + apply + residual
  bn_stats<<<1, CCH, 0, stream>>>(sums, gamma, beta, ssb);
  bn_apply<<<(NEL / 4) / 256, 256, 0, stream>>>(yb, ssb, x, (float*)d_out,
                                                NEL / 4);
}
